// PairmixLayer_71966472012429
// MI455X (gfx1250) — matrix operand order repaired and checked by reading, not compiled, not run on hardware
//
#include <hip/hip_runtime.h>

typedef __attribute__((ext_vector_type(16))) _Float16 v16h;
typedef __attribute__((ext_vector_type(8)))  float    v8f;
typedef __attribute__((ext_vector_type(4)))  float    v4f;

#define NPAIRS   40000
#define NFEAT    128
#define CHANS    15
#define CHSTRIDE (NPAIRS * NFEAT)   // 5,120,000 elements per (l,m) channel
#define NTILE    16

__global__ __launch_bounds__(256) void pairmix_fused(
    const float* __restrict__ x, const float* __restrict__ y,
    const float* __restrict__ g, const float* __restrict__ w1,
    const float* __restrict__ w2, const float* __restrict__ cg,
    float* __restrict__ out)
{
  __shared__ _Float16 h16[NTILE][NFEAT];   // Hermite features, f16 for WMMA A
  __shared__ float    u_lds[NTILE][NFEAT]; // u = w1g * w2g
  __shared__ float    cg_lds[3375];        // full CG tensor (13.5 KB)

  const int t  = threadIdx.x;
  const int n0 = blockIdx.x * NTILE;

  // ---- CG tensor -> LDS (all threads) ----
  for (int i = t; i < 3375; i += 256) cg_lds[i] = cg[i];

  // ---- Stage 0: Hermite recurrence for 16 pairs (threads 0..15) ----
  if (t < NTILE) {
    const float gv = g[n0 + t];
    float hm = 0.7511255444649425f * expf(-0.5f * gv * gv);  // pi^-1/4 * exp(-g^2/2)
    float hk = 1.4142135623730951f * gv * hm;                // sqrt(2)*g*h0
    h16[t][0] = (_Float16)hm;
    h16[t][1] = (_Float16)hk;
    for (int k = 1; k < NFEAT - 1; ++k) {
      const float kf = (float)k;
      const float hn = sqrtf(2.0f / (kf + 1.0f)) * gv * hk
                     - sqrtf(kf / (kf + 1.0f)) * hm;
      h16[t][k + 1] = (_Float16)hn;
      hm = hk; hk = hn;
    }
  }
  __syncthreads();

  // ---- Stage 1: WMMA GEMMs w1g = h*w1^T, w2g = h*w2^T; u = w1g*w2g ----
  // 8 waves, wave w owns the 16n x 16f tile with f in [16w, 16w+16).
  {
    const int  lane = t & 31;
    const int  wv   = t >> 5;
    const int  row  = lane & 15;        // A: matrix row (= local pair); D: column
    const bool hi   = lane >= 16;
    const int  fcol = wv * 16 + row;    // this lane's output feature column
    v8f acc1 = {}; v8f acc2 = {};
#pragma unroll
    for (int kt = 0; kt < 4; ++kt) {    // K = 128 in 4 steps of 32
      const int k0 = kt * 32;
      v16h a, b1, b2;
#pragma unroll
      for (int j = 0; j < 16; ++j) {
        // A 16x32 f16, contiguous k order used here (same as B): lane l -> M = l%16; K = (l>=16?16:0) + j
        const int ka = k0 + (hi ? 16 : 0) + j;
        a[j] = h16[row][ka];
        // B 32x16 f16, contiguous k order used here (same as A): lane l -> N = l%16; K = (l>=16?16:0) + j
        const int kb = k0 + (hi ? 16 : 0) + j;
        b1[j] = (_Float16)w1[fcol * NFEAT + kb];   // B[k][f] = w1[f][k]
        b2[j] = (_Float16)w2[fcol * NFEAT + kb];
      }
      acc1 = __builtin_amdgcn_wmma_f32_16x16x32_f16(false, a, false, b1,
                                                    (short)0, acc1, false, false);
      acc2 = __builtin_amdgcn_wmma_f32_16x16x32_f16(false, a, false, b2,
                                                    (short)0, acc2, false, false);
    }
    // D layout: lane l -> col = l%16; VGPR r -> row = r + (l>=16 ? 8 : 0)
#pragma unroll
    for (int r = 0; r < 8; ++r) {
      const int nrow = r + (hi ? 8 : 0);
      u_lds[nrow][fcol] = acc1[r] * acc2[r];
    }
  }
  __syncthreads();

  // ---- Stage 2: CG contraction, float4-vectorized, non-temporal streams ----
  // thread t -> feature quad fq = t%32 (wave covers 512 contiguous bytes),
  // pair rows: n = (t/32) + 8*it, it = 0..1 -> all 16 local pairs covered.
  const int fq    = t & 31;          // quad index; f = 4*fq
  const int nbase = t >> 5;          // 0..7
#pragma unroll 1
  for (int it = 0; it < 2; ++it) {
    const int    nl   = nbase + 8 * it;
    const size_t base = (size_t)(n0 + nl) * NFEAT + 4 * fq;
    v4f xv[CHANS], yv[CHANS], ov[CHANS];
#pragma unroll
    for (int c = 0; c < CHANS; ++c) {
      xv[c] = __builtin_nontemporal_load((const v4f*)(x + base + (size_t)c * CHSTRIDE));
      yv[c] = __builtin_nontemporal_load((const v4f*)(y + base + (size_t)c * CHSTRIDE));
      ov[c] = (v4f){0.0f, 0.0f, 0.0f, 0.0f};
    }
#pragma unroll
    for (int l3 = 0; l3 < 3; ++l3) {
#pragma unroll
      for (int l1 = 0; l1 < 3; ++l1) {
#pragma unroll
        for (int l2 = 0; l2 < 3; ++l2) {
          const int lo = (l1 > l2) ? (l1 - l2) : (l2 - l1);
          if (l3 < lo || l3 > l1 + l2) continue;   // triangle rule (const-folded)
#pragma unroll
          for (int m1 = -l1; m1 <= l1; ++m1) {
#pragma unroll
            for (int m2 = -l2; m2 <= l2; ++m2) {
              const int m3 = m1 + m2;
              if (m3 < -l3 || m3 > l3) continue;   // m-selection (const-folded)
              const int c1 = l1 * 5 + m1 + 2;
              const int c2 = l2 * 5 + m2 + 2;
              const int c3 = l3 * 5 + m3 + 2;
              const int ci = ((((l3 * 5 + (m3 + 2)) * 3 + l1) * 5 + (m1 + 2)) * 3
                              + l2) * 5 + (m2 + 2);
              const float cgv = cg_lds[ci];
              ov[c3] += cgv * (xv[c1] * yv[c2] + yv[c1] * xv[c2]);
            }
          }
        }
      }
    }
    const v4f uu = *(const v4f*)&u_lds[nl][4 * fq];
#pragma unroll
    for (int c = 0; c < CHANS; ++c) {
      const v4f r = ov[c] * uu;
      __builtin_nontemporal_store(r, (v4f*)(out + base + (size_t)c * CHSTRIDE));
    }
  }
}

extern "C" void kernel_launch(void* const* d_in, const int* in_sizes, int n_in,
                              void* d_out, int out_size, void* d_ws, size_t ws_size,
                              hipStream_t stream) {
  const float* x  = (const float*)d_in[0];
  const float* y  = (const float*)d_in[1];
  const float* g  = (const float*)d_in[2];
  const float* w1 = (const float*)d_in[3];
  const float* w2 = (const float*)d_in[4];
  const float* cg = (const float*)d_in[5];
  float* out = (float*)d_out;

  dim3 grid(NPAIRS / NTILE);   // 2500 blocks, 16 pairs each (40000 % 16 == 0)
  dim3 block(256);             // 8 wave32 waves
  hipLaunchKernelGGL(pairmix_fused, grid, block, 0, stream,
                     x, y, g, w1, w2, cg, out);
}
